// AggDeepAttnMISL_36988258353163
// MI455X (gfx1250) — compile-verified, matrix operand order checked by reading, not run on hardware
//
#include <hip/hip_runtime.h>

#define N_ 4
#define P_ 20000
#define DIN 1024
#define K_ 8
#define DH 64
#define KCHUNK 256
#define BSTRIDE 272   // bf16 elems per Bt row: 256 + 16 pad (544B, 16B-aligned, bank-skewed)
#define NCOLS 80      // 64 proj cols + 8 centroid cols + 8 zero pad

typedef __bf16 v16bf __attribute__((ext_vector_type(16)));
typedef __bf16 v8bf  __attribute__((ext_vector_type(8)));
typedef float  v8f   __attribute__((ext_vector_type(8)));

__global__ void __launch_bounds__(256) zero_ws_kernel(float* ws, int n) {
  int i = blockIdx.x * 256 + threadIdx.x;
  if (i < n) ws[i] = 0.0f;
}

// Fused: bf16 WMMA GEMM  xp = relu(x@W_pre + b_pre)  AND  dots = x@centroids^T,
// then hard cluster assignment + per-WG pooling accumulation.
__global__ void __launch_bounds__(256) fused_proj_assign_kernel(
    const float* __restrict__ x, const float* __restrict__ centroids,
    const float* __restrict__ W_pre, const float* __restrict__ b_pre,
    float* __restrict__ accG, float* __restrict__ cntG,
    float* __restrict__ out1, float* __restrict__ out2)
{
  __shared__ __align__(16) __bf16 Bt[NCOLS][BSTRIDE]; // B^T chunk: [col][k-in-chunk]
  __shared__ float accF[N_][K_][DH];
  __shared__ float cntF[N_][K_];
  __shared__ float c2s[K_];
  __shared__ float dotS[8][16][K_];
  __shared__ float x2S[8][16];
  __shared__ int   asgS[8][16];

  const int tid   = threadIdx.x;
  const int wid   = tid >> 5;
  const int lane  = tid & 31;
  const int col16 = lane & 15;
  const int hi    = lane >> 4;

  // Per-wave tile: 5000 tiles total, 1250 per n (20000 % 16 == 0 -> no straddle)
  const int  g     = blockIdx.x * 8 + wid;
  const int  n     = g / (P_ / 16);
  const long gRow0 = (long)g * 16;

  // ---- init LDS accumulators; compute centroid squared norms (wave w -> centroid w)
  for (int i = tid; i < N_ * K_ * DH; i += 256) ((float*)accF)[i] = 0.0f;
  for (int i = tid; i < N_ * K_;      i += 256) ((float*)cntF)[i] = 0.0f;
  {
    float part = 0.0f;
    const float* cp = centroids + (long)wid * DIN;
    for (int d = lane; d < DIN; d += 32) { float v = cp[d]; part += v * v; }
    #pragma unroll
    for (int off = 16; off; off >>= 1) part += __shfl_down(part, off, 32);
    if (lane == 0) c2s[wid] = part;
  }

  v8f acc[5];
  #pragma unroll
  for (int t = 0; t < 5; t++) {
    #pragma unroll
    for (int e = 0; e < 8; e++) acc[t][e] = 0.0f;
  }
  float x2part = 0.0f;
  const float* xrow = x + (gRow0 + col16) * (long)DIN;

  for (int kc = 0; kc < DIN; kc += KCHUNK) {
    __syncthreads();   // previous chunk's compute done (also covers init phase)
    // stage W_pre chunk: Bt[col][kk] = W_pre[kc+kk][col]   (coalesced: 64 cols/row)
    {
      int col = tid & 63, kk0 = tid >> 6;
      for (int kk = kk0; kk < KCHUNK; kk += 4)
        Bt[col][kk] = (__bf16)W_pre[(long)(kc + kk) * DH + col];
    }
    // stage centroids^T into cols 64..71, zero cols 72..79
    {
      const float* cp = centroids + (long)wid * DIN + kc;
      for (int kk = lane; kk < KCHUNK; kk += 32) {
        Bt[64 + wid][kk] = (__bf16)cp[kk];
        Bt[72 + wid][kk] = (__bf16)0.0f;
      }
    }
    __syncthreads();

    #pragma unroll
    for (int s = 0; s < KCHUNK / 32; s++) {
      // A fragment (16x32 bf16): lane row = lane&15; hi-half picks K groups.
      // K runs: [kb, kb+7] and [kb+16, kb+23] with kb = base + hi*8  (ISA A layout)
      const int kb = kc + s * 32 + hi * 8;
      float4 a0 = *(const float4*)(xrow + kb);
      float4 a1 = *(const float4*)(xrow + kb + 4);
      float4 a2 = *(const float4*)(xrow + kb + 16);
      float4 a3 = *(const float4*)(xrow + kb + 20);
      x2part += a0.x*a0.x + a0.y*a0.y + a0.z*a0.z + a0.w*a0.w
              + a1.x*a1.x + a1.y*a1.y + a1.z*a1.z + a1.w*a1.w
              + a2.x*a2.x + a2.y*a2.y + a2.z*a2.z + a2.w*a2.w
              + a3.x*a3.x + a3.y*a3.y + a3.z*a3.z + a3.w*a3.w;
      v16bf a;
      a[0]=(__bf16)a0.x; a[1]=(__bf16)a0.y; a[2]=(__bf16)a0.z; a[3]=(__bf16)a0.w;
      a[4]=(__bf16)a1.x; a[5]=(__bf16)a1.y; a[6]=(__bf16)a1.z; a[7]=(__bf16)a1.w;
      a[8]=(__bf16)a2.x; a[9]=(__bf16)a2.y; a[10]=(__bf16)a2.z; a[11]=(__bf16)a2.w;
      a[12]=(__bf16)a3.x; a[13]=(__bf16)a3.y; a[14]=(__bf16)a3.z; a[15]=(__bf16)a3.w;

      const int klocal = s * 32 + hi * 8;
      #pragma unroll
      for (int t = 0; t < 5; t++) {
        const __bf16* bp = &Bt[t * 16 + col16][klocal];
        v8bf b0 = *(const v8bf*)(bp);
        v8bf b1 = *(const v8bf*)(bp + 16);
        v16bf b;
        #pragma unroll
        for (int e = 0; e < 8; e++) { b[e] = b0[e]; b[8 + e] = b1[e]; }
        acc[t] = __builtin_amdgcn_wmma_f32_16x16x32_bf16(
            false, a, false, b, (short)0, acc[t], false, false);
      }
    }
  }

  // ---- epilogue: ||x||^2 per row, dots -> LDS, per-row hard assignment
  float x2row = x2part + __shfl_xor(x2part, 16, 32);
  if (hi == 0) x2S[wid][col16] = x2row;
  #pragma unroll
  for (int r = 0; r < 8; r++)
    if (col16 < K_) dotS[wid][r + 8 * hi][col16] = acc[4][r];
  __syncthreads();

  if (lane < 16) {
    float x2r = x2S[wid][lane];
    float l[K_], m = -3.0e38f;
    #pragma unroll
    for (int k = 0; k < K_; k++) {
      float d2 = x2r + c2s[k] - 2.0f * dotS[wid][lane][k];
      float d  = sqrtf(fmaxf(d2, 0.0f));
      d = fmaxf(d, 1e-20f);
      l[k] = 1.0e7f / d;
      m = fmaxf(m, l[k]);
    }
    float e[K_], ssum = 0.0f;
    #pragma unroll
    for (int k = 0; k < K_; k++) { e[k] = __expf(l[k] - m); ssum += e[k]; }
    int a = -1;
    #pragma unroll
    for (int k = 0; k < K_; k++) if (e[k] > 0.5f * ssum) a = k;
    asgS[wid][lane] = a;
    if (a >= 0) atomicAdd(&cntF[n][a], 1.0f);
  }
  __syncthreads();

  // ---- bias + relu, store enc_seq (twice), pool into LDS cluster accumulators
  float bpc[4];
  #pragma unroll
  for (int t = 0; t < 4; t++) bpc[t] = b_pre[t * 16 + col16];
  #pragma unroll
  for (int t = 0; t < 4; t++) {
    #pragma unroll
    for (int r = 0; r < 8; r++) {
      const int row = r + 8 * hi;
      float v = fmaxf(acc[t][r] + bpc[t], 0.0f);
      const long o = (gRow0 + row) * (long)DH + t * 16 + col16;
      out1[o] = v;
      out2[o] = v;
      int a = asgS[wid][row];
      if (a >= 0) atomicAdd(&accF[n][a][t * 16 + col16], v);
    }
  }
  __syncthreads();

  // ---- hierarchical flush to global accumulators
  for (int i = tid; i < N_ * K_ * DH; i += 256) atomicAdd(&accG[i], ((float*)accF)[i]);
  for (int i = tid; i < N_ * K_;      i += 256) atomicAdd(&cntG[i], ((float*)cntF)[i]);
}

// Tiny finalize: normalize cluster sums, attention MLP, softmax over K, project.
__global__ void __launch_bounds__(64) finalize_kernel(
    const float* __restrict__ accG, const float* __restrict__ cntG,
    const float* __restrict__ W_a1, const float* __restrict__ b_a1,
    const float* __restrict__ W_a2, const float* __restrict__ b_a2,
    const float* __restrict__ W_out, const float* __restrict__ b_out,
    float* __restrict__ enc_cls)
{
  const int n = blockIdx.x, t = threadIdx.x;
  __shared__ float xc[K_][DH];
  __shared__ float Avals[K_];
  __shared__ float wa[K_];
  __shared__ float pooled[DH];

  for (int i = t; i < K_ * DH; i += 64) {
    int k = i / DH, d = i % DH;
    float c = cntG[n * K_ + k];
    xc[k][d] = (c > 0.0f) ? accG[(n * K_ + k) * DH + d] / c : 0.0f;
  }
  __syncthreads();

  if (t < K_) {
    float s2 = b_a2[0];
    for (int j = 0; j < 32; j++) {
      float h = b_a1[j];
      for (int d = 0; d < DH; d++) h += xc[t][d] * W_a1[d * 32 + j];
      s2 += tanhf(h) * W_a2[j];
    }
    float c = cntG[n * K_ + t];
    Avals[t] = (c > 0.0f) ? s2 : -100000.0f;
  }
  __syncthreads();

  if (t == 0) {
    float m = Avals[0];
    for (int k = 1; k < K_; k++) m = fmaxf(m, Avals[k]);
    float s = 0.0f;
    for (int k = 0; k < K_; k++) { wa[k] = __expf(Avals[k] - m); s += wa[k]; }
    for (int k = 0; k < K_; k++) wa[k] /= s;
  }
  __syncthreads();

  if (t < DH) {
    float p = 0.0f;
    for (int k = 0; k < K_; k++) p += xc[k][t] * wa[k];
    pooled[t] = p;
  }
  __syncthreads();

  if (t < 32) {
    float s = b_out[t];
    for (int d = 0; d < DH; d++) s += pooled[d] * W_out[d * 32 + t];
    enc_cls[n * 32 + t] = fmaxf(s, 0.0f);
  }
}

extern "C" void kernel_launch(void* const* d_in, const int* in_sizes, int n_in,
                              void* d_out, int out_size, void* d_ws, size_t ws_size,
                              hipStream_t stream) {
  const float* x         = (const float*)d_in[0];
  const float* centroids = (const float*)d_in[1];
  const float* W_pre     = (const float*)d_in[2];
  const float* b_pre     = (const float*)d_in[3];
  const float* W_a1      = (const float*)d_in[4];
  const float* b_a1      = (const float*)d_in[5];
  const float* W_a2      = (const float*)d_in[6];
  const float* b_a2      = (const float*)d_in[7];
  const float* W_out     = (const float*)d_in[8];
  const float* b_out     = (const float*)d_in[9];

  float* out     = (float*)d_out;
  float* enc_cls = out;                                  // [4,32]
  float* seq1    = out + N_ * 32;                        // [80000,64]
  float* seq2    = seq1 + (long)N_ * P_ * DH;            // [80000,64] (duplicate)

  float* accG = (float*)d_ws;                            // [4][8][64]
  float* cntG = accG + N_ * K_ * DH;                     // [4][8]
  const int nacc = N_ * K_ * DH + N_ * K_;

  zero_ws_kernel<<<(nacc + 255) / 256, 256, 0, stream>>>(accG, nacc);
  fused_proj_assign_kernel<<<(N_ * P_ / 16) / 8, 256, 0, stream>>>(
      x, centroids, W_pre, b_pre, accG, cntG, seq1, seq2);
  finalize_kernel<<<N_, 64, 0, stream>>>(
      accG, cntG, W_a1, b_a1, W_a2, b_a2, W_out, b_out, enc_cls);
}
